// SelfAttention2D_old_4672924418533
// MI455X (gfx1250) — matrix operand order repaired and checked by reading, not compiled, not run on hardware
//
#include <hip/hip_runtime.h>
#include <hip/hip_bf16.h>

#define BATCH 8
#define NTOK  4096
#define CCH   128
#define DD    16

typedef __attribute__((ext_vector_type(16))) __bf16       v16bf;
typedef __attribute__((ext_vector_type(8)))  float        v8f;
typedef __attribute__((ext_vector_type(4)))  float        f32x4;
typedef __attribute__((ext_vector_type(4)))  unsigned int u32x4;

// A/B operand register block for V_WMMA_*_16X16X32_BF16: 16 bf16 = 8 VGPRs.
union ABReg {
    v16bf          v;
    unsigned short h[16];
    u32x4          q[2];
};

__device__ __forceinline__ unsigned short f2bfu(float f) {
    union { __bf16 b; unsigned short s; } u;
    u.b = (__bf16)f;             // RNE convert, lowers to v_cvt on gfx1250
    return u.s;
}

// ---------------------------------------------------------------------------
// Kernel A: f = x@w_f, g = x@w_g, hh = x@w_h  (bf16 outputs)
//   fq, gk stored row-major [b][n][16]; hh stored transposed hvT[b][d][n]
//   so later B-operand loads are contiguous.
// One wave = one 16-row M tile, K=128 via 4 x (16x16x32) WMMA.
// ---------------------------------------------------------------------------
__global__ void __launch_bounds__(128)
proj_kernel(const float* __restrict__ x,
            const float* __restrict__ wf, const float* __restrict__ wg,
            const float* __restrict__ wh,
            unsigned short* __restrict__ fq, unsigned short* __restrict__ gk,
            unsigned short* __restrict__ hvT) {
    const int lane = threadIdx.x & 31;
    const int wave = threadIdx.x >> 5;
    const int b    = blockIdx.x >> 6;         // 64 blocks per batch
    const int rb   = blockIdx.x & 63;
    const int n0   = rb * 64 + wave * 16;
    const int col  = lane & 15;               // M row (A) / N col (B,D)
    const int half = lane >> 4;

    v8f accF = {}, accG = {}, accH = {};
    const float* xrow = x + ((size_t)b * NTOK + (n0 + col)) * CCH;

    for (int kk = 0; kk < CCH; kk += 32) {
        // A operand: lane holds row n0+col, K [kk+16h, kk+16h+16)
        ABReg a;
        const float* p1 = xrow + kk + 16 * half;
        const float* p2 = xrow + kk + 16 * half + 8;
        f32x4 c1a = *(const f32x4*)(p1);
        f32x4 c1b = *(const f32x4*)(p1 + 4);
        f32x4 c2a = *(const f32x4*)(p2);
        f32x4 c2b = *(const f32x4*)(p2 + 4);
#pragma unroll
        for (int j = 0; j < 4; ++j) {
            a.h[j]      = f2bfu(c1a[j]);
            a.h[4 + j]  = f2bfu(c1b[j]);
            a.h[8 + j]  = f2bfu(c2a[j]);
            a.h[12 + j] = f2bfu(c2b[j]);
        }
        // B operands: lane holds column d=col, K rows kk+16h+j (weights are
        // tiny: 24 KB total, L2/L0 resident).
        ABReg bF, bG, bH;
#pragma unroll
        for (int j = 0; j < 16; ++j) {
            const int k = kk + 16 * half + j;
            bF.h[j] = f2bfu(wf[k * DD + col]);
            bG.h[j] = f2bfu(wg[k * DD + col]);
            bH.h[j] = f2bfu(wh[k * DD + col]);
        }
        accF = __builtin_amdgcn_wmma_f32_16x16x32_bf16(false, a.v, false, bF.v,
                                                       (short)0, accF, false, false);
        accG = __builtin_amdgcn_wmma_f32_16x16x32_bf16(false, a.v, false, bG.v,
                                                       (short)0, accG, false, false);
        accH = __builtin_amdgcn_wmma_f32_16x16x32_bf16(false, a.v, false, bH.v,
                                                       (short)0, accH, false, false);
    }

    // D layout: row = i + 8*half, col = lane%16
#pragma unroll
    for (int i = 0; i < 8; ++i) {
        const int n = n0 + i + 8 * half;
        fq [((size_t)b * NTOK + n) * DD + col]  = f2bfu(accF[i]);
        gk [((size_t)b * NTOK + n) * DD + col]  = f2bfu(accG[i]);
        hvT[((size_t)b * DD + col) * NTOK + n]  = f2bfu(accH[i]);
    }
}

// ---------------------------------------------------------------------------
// Kernel B: per-column softmax stats over axis n (the reference's axis=1):
//   M[m] = max_n f[n].g[m],  Z[m] = sum_n exp(s[n,m]-M[m])
// One wave owns one 16-column m-tile; loops all 256 n-tiles with WMMA,
// online max/sum; lanes (L, L+16) hold the same column -> shuffle-combine.
// ---------------------------------------------------------------------------
__global__ void __launch_bounds__(256)
colstats_kernel(const unsigned short* __restrict__ fq,
                const unsigned short* __restrict__ gk,
                float* __restrict__ colmax, float* __restrict__ colsum) {
    const int lane = threadIdx.x & 31;
    const int wave = threadIdx.x >> 5;
    const int b    = blockIdx.x >> 5;         // 32 blocks per batch
    const int mb   = blockIdx.x & 31;
    const int m0   = mb * 128 + wave * 16;
    const int col  = lane & 15;
    const int half = lane >> 4;

    // B operand = g tile transposed (K = d = 16, pad to 32): lane half h holds d [8h,8h+8), elements 8..15 zero.
    ABReg bg = {};
    {
        const unsigned short* grow = gk + ((size_t)b * NTOK + (m0 + col)) * DD;
        bg.q[0] = *(const u32x4*)(grow + 8 * half);
    }

    float mx = -3.0e38f, sm = 0.0f;
    const unsigned short* fbase = fq + (size_t)b * NTOK * DD;

    for (int nt = 0; nt < NTOK / 16; ++nt) {
        ABReg a = {};
        a.q[0] = *(const u32x4*)(fbase + (size_t)(nt * 16 + col) * DD + 8 * half);
        v8f z = {};
        v8f s = __builtin_amdgcn_wmma_f32_16x16x32_bf16(false, a.v, false, bg.v,
                                                        (short)0, z, false, false);
        float tm = fmaxf(fmaxf(fmaxf(s[0], s[1]), fmaxf(s[2], s[3])),
                         fmaxf(fmaxf(s[4], s[5]), fmaxf(s[6], s[7])));
        float nm  = fmaxf(mx, tm);
        float acc = 0.0f;
#pragma unroll
        for (int i = 0; i < 8; ++i) acc += __expf(s[i] - nm);
        sm = sm * __expf(mx - nm) + acc;
        mx = nm;
    }
    // combine the two half-rows (same column m, disjoint n rows)
    float omx = __shfl_xor(mx, 16);
    float osm = __shfl_xor(sm, 16);
    float nm  = fmaxf(mx, omx);
    float tot = sm * __expf(mx - nm) + osm * __expf(omx - nm);
    if (half == 0) {
        colmax[(size_t)b * NTOK + m0 + col] = nm;
        colsum[(size_t)b * NTOK + m0 + col] = tot;
    }
}

// ---------------------------------------------------------------------------
// Kernel C: y[n] = sum_m exp(f[n].g[m]-M[m])/Z[m] * hh[m]; out = gamma*(y@w_v)+x
// One wave owns one 16-row n-tile. m processed 32 at a time: two K16-padded
// score WMMAs feed one full-K32 y WMMA. P tile transposed C-layout -> A-layout
// through a per-wave LDS tile (in-order DS within a wave, no barrier).
// ---------------------------------------------------------------------------
__global__ void __launch_bounds__(256)
attn_out_kernel(const float* __restrict__ x, const float* __restrict__ wv,
                const float* __restrict__ gamma,
                const unsigned short* __restrict__ fq,
                const unsigned short* __restrict__ gk,
                const unsigned short* __restrict__ hvT,
                const float* __restrict__ colmax, const float* __restrict__ colsum,
                float* __restrict__ out) {
    __shared__ __align__(16) unsigned short tile[8][16 * 32];  // 16 rows x 32 K, per wave

    const int lane = threadIdx.x & 31;
    const int wave = threadIdx.x >> 5;
    const int b    = blockIdx.x >> 5;
    const int nb   = blockIdx.x & 31;
    const int n0   = nb * 128 + wave * 16;
    const int col  = lane & 15;
    const int half = lane >> 4;

    // Loop-invariant A operand: f rows n0..n0+15 (K = d = 16, padded).
    ABReg af = {};
    af.q[0] = *(const u32x4*)(fq + ((size_t)b * NTOK + (n0 + col)) * DD + 8 * half);

    const float gmm = gamma[0];
    v8f yacc = {};

    const unsigned short* gbase = gk  + (size_t)b * NTOK * DD;
    const unsigned short* hbase = hvT + (size_t)b * DD * NTOK;
    const float* cmb = colmax + (size_t)b * NTOK;
    const float* csb = colsum + (size_t)b * NTOK;
    unsigned short* myTile = &tile[wave][0];

    for (int mt = 0; mt < NTOK; mt += 32) {
        if (mt + 32 < NTOK) {   // prefetch next g tile -> global_prefetch_b8
            __builtin_prefetch(gbase + (size_t)(mt + 32 + col) * DD, 0, 1);
        }
        ABReg bg0 = {}, bg1 = {};
        {
            const unsigned short* g0 = gbase + (size_t)(mt + col) * DD;
            const unsigned short* g1 = gbase + (size_t)(mt + 16 + col) * DD;
            bg0.q[0] = *(const u32x4*)(g0 + 8 * half);
            bg1.q[0] = *(const u32x4*)(g1 + 8 * half);
        }
        v8f z = {};
        v8f s0 = __builtin_amdgcn_wmma_f32_16x16x32_bf16(false, af.v, false, bg0.v,
                                                         (short)0, z, false, false);
        v8f s1 = __builtin_amdgcn_wmma_f32_16x16x32_bf16(false, af.v, false, bg1.v,
                                                         (short)0, z, false, false);
        const float cm0 = cmb[mt + col],      cm1 = cmb[mt + 16 + col];
        const float rz0 = __builtin_amdgcn_rcpf(csb[mt + col]);
        const float rz1 = __builtin_amdgcn_rcpf(csb[mt + 16 + col]);

        // P tile (16 n-rows x 32 m-cols) -> LDS, then reload in A layout
#pragma unroll
        for (int i = 0; i < 8; ++i) {
            const int r = i + 8 * half;
            myTile[r * 32 + col]      = f2bfu(__expf(s0[i] - cm0) * rz0);
            myTile[r * 32 + 16 + col] = f2bfu(__expf(s1[i] - cm1) * rz1);
        }
        ABReg ap;
        ap.q[0] = *(const u32x4*)(myTile + col * 32 + 8 * half);        // K [8h,8h+8)
        ap.q[1] = *(const u32x4*)(myTile + col * 32 + 16 + 8 * half);   // K [16+8h,..)

        // B operand: hh tile, full K=32 (contiguous along n thanks to hvT layout)
        ABReg bh;
        const unsigned short* hr = hbase + (size_t)col * NTOK + mt + 8 * half;
        bh.q[0] = *(const u32x4*)(hr);
        bh.q[1] = *(const u32x4*)(hr + 16);

        yacc = __builtin_amdgcn_wmma_f32_16x16x32_bf16(false, ap.v, false, bh.v,
                                                       (short)0, yacc, false, false);
    }

    // Output projection: out = gamma * (y @ w_v) + x. Transpose y through LDS.
#pragma unroll
    for (int i = 0; i < 8; ++i)
        myTile[(i + 8 * half) * 16 + col] = f2bfu(yacc[i]);
    ABReg ay = {};
    ay.q[0] = *(const u32x4*)(myTile + col * 16 + 8 * half);

    for (int ct = 0; ct < 8; ++ct) {
        const int c0 = ct * 16;
        ABReg bv = {};
        {
#pragma unroll
            for (int k = 0; k < 8; ++k)
                bv.h[k] = f2bfu(wv[(8 * half + k) * CCH + c0 + col]);
        }
        v8f z = {};
        v8f o = __builtin_amdgcn_wmma_f32_16x16x32_bf16(false, ay.v, false, bv.v,
                                                        (short)0, z, false, false);
#pragma unroll
        for (int i = 0; i < 8; ++i) {
            const size_t idx = ((size_t)b * NTOK + (n0 + i + 8 * half)) * CCH + c0 + col;
            out[idx] = gmm * o[i] + x[idx];
        }
    }
}

// ---------------------------------------------------------------------------
extern "C" void kernel_launch(void* const* d_in, const int* in_sizes, int n_in,
                              void* d_out, int out_size, void* d_ws, size_t ws_size,
                              hipStream_t stream) {
    (void)in_sizes; (void)n_in; (void)out_size; (void)ws_size;
    const float* x     = (const float*)d_in[0];
    const float* wf    = (const float*)d_in[1];
    const float* wg    = (const float*)d_in[2];
    const float* wh    = (const float*)d_in[3];
    const float* wv    = (const float*)d_in[4];
    const float* gamma = (const float*)d_in[5];

    char* ws = (char*)d_ws;
    const size_t SZ_BF = (size_t)BATCH * NTOK * DD * sizeof(unsigned short); // 1 MB
    unsigned short* fq  = (unsigned short*)(ws);
    unsigned short* gk  = (unsigned short*)(ws + SZ_BF);
    unsigned short* hvT = (unsigned short*)(ws + 2 * SZ_BF);
    float* colmax       = (float*)(ws + 3 * SZ_BF);
    float* colsum       = (float*)(ws + 3 * SZ_BF + (size_t)BATCH * NTOK * sizeof(float));

    proj_kernel<<<dim3(BATCH * 64), dim3(128), 0, stream>>>(x, wf, wg, wh, fq, gk, hvT);
    colstats_kernel<<<dim3(BATCH * 32), dim3(256), 0, stream>>>(fq, gk, colmax, colsum);
    attn_out_kernel<<<dim3(BATCH * 32), dim3(256), 0, stream>>>(x, wv, gamma, fq, gk, hvT,
                                                                colmax, colsum, (float*)d_out);
}
